// CrossAttention_17927193494170
// MI455X (gfx1250) — matrix operand order repaired and checked by reading, not compiled, not run on hardware
//
#include <hip/hip_runtime.h>

// ---------------------------------------------------------------------------
// CDNA5 (gfx1250, wave32) cross-attention pipeline.
// Matmuls: v_wmma_f32_16x16x32_f16 (f16 in, f32 accumulate).
// Data movement: global_load_async_to_lds_b128 (ASYNCcnt) double-buffered.
// ---------------------------------------------------------------------------

typedef _Float16 v16h __attribute__((ext_vector_type(16)));
typedef float    v8f  __attribute__((ext_vector_type(8)));
typedef _Float16 h4   __attribute__((ext_vector_type(4)));
typedef int      v4i_v __attribute__((vector_size(16)));   // matches builtin's V4i

#define AS1 __attribute__((address_space(1)))
#define AS3 __attribute__((address_space(3)))

union U16h { v16h v; uint4 q[2]; };

// Async 16-byte global -> LDS copy (CDNA5 ASYNCcnt path), with sync fallback.
__device__ __forceinline__ void async_cp16(void* lds, const void* g) {
#if __has_builtin(__builtin_amdgcn_global_load_async_to_lds_b128)
  __builtin_amdgcn_global_load_async_to_lds_b128((AS1 v4i_v*)(g), (AS3 v4i_v*)(lds), 0, 0);
#else
  *(uint4*)lds = *(const uint4*)g;
#endif
}

__device__ __forceinline__ void async_wait0() {
#if __has_builtin(__builtin_amdgcn_s_wait_asynccnt)
  __builtin_amdgcn_s_wait_asynccnt(0);
#else
  __asm__ volatile("s_wait_asynccnt 0x0" ::: "memory");
#endif
}

// A-matrix fragment (16x32, MxK, f16). Wave32 layout per CDNA5 ISA 7.12.2:
// lane m = lane&15; lanes 0-15 hold K=0..7 & 16..23, lanes 16-31 K=8..15 & 24..31.
__device__ __forceinline__ v16h load_afrag(const _Float16* base, long row0,
                                           int k0, int ld, int lane) {
  int m  = lane & 15;
  int kb = (lane >> 4) * 8;
  const _Float16* p = base + (row0 + m) * (long)ld + k0 + kb;
  U16h u;
  u.q[0] = *(const uint4*)(p);        // K = kb .. kb+7
  u.q[1] = *(const uint4*)(p + 16);   // K = kb+16 .. kb+23
  return u.v;
}

// B-matrix fragment (32x16, KxN, f16) from row-major src with src row = N,
// src col = K (B[k][n] = src[row0+n][k0+k]); lane n = lane&15,
// lanes 0-15 hold K=0..7 & 16..23, lanes 16-31 K=8..15 & 24..31 (matches A).
__device__ __forceinline__ v16h load_bfrag(const _Float16* base, long row0,
                                           int k0, int ld, int lane) {
  int n   = lane & 15;
  int klo = (lane >> 4) * 8;
  const _Float16* p = base + (row0 + n) * (long)ld + k0 + klo;
  U16h u;
  u.q[0] = *(const uint4*)(p);
  u.q[1] = *(const uint4*)(p + 16);
  return u.v;
}

__device__ __forceinline__ v8f wmma16(v16h a, v16h b, v8f c) {
  return __builtin_amdgcn_wmma_f32_16x16x32_f16(false, a, false, b,
                                                (short)0, c, false, false);
}

// ---------------------------------------------------------------------------
// f32 -> f16 conversion (vectorized, grid-stride). n must be a multiple of 4.
// ---------------------------------------------------------------------------
__global__ __launch_bounds__(256) void cvt_f32_f16(const float* __restrict__ s,
                                                   _Float16* __restrict__ d,
                                                   long n) {
  long i  = ((long)blockIdx.x * blockDim.x + threadIdx.x) * 4;
  long st = (long)gridDim.x * blockDim.x * 4;
  for (; i < n; i += st) {
    float4 f = *(const float4*)(s + i);
    h4 h;
    h.x = (_Float16)f.x; h.y = (_Float16)f.y;
    h.z = (_Float16)f.z; h.w = (_Float16)f.w;
    *(h4*)(d + i) = h;
  }
}

// ---------------------------------------------------------------------------
// WMMA GEMM: C[M,N] = A[M,K] * W[N,K]^T  (nn.Linear layout).
// 256 thr = 8 waves (2x4), block tile 64x128, wave tile 32x32, K-step 32.
// Double-buffered async global->LDS staging (A: 4KB, W: 8KB per step).
// LDS rows padded to ld=40 halfs (80B) for conflict-free 16-lane reads.
// ---------------------------------------------------------------------------
template <bool HALF_OUT>
__global__ __launch_bounds__(256) void gemm_wmma(const _Float16* __restrict__ A,
                                                 const _Float16* __restrict__ W,
                                                 float* __restrict__ Cf,
                                                 _Float16* __restrict__ Ch,
                                                 int M, int N, int K,
                                                 float outScale) {
  __shared__ alignas(16) _Float16 Ash[2][64 * 40];
  __shared__ alignas(16) _Float16 Wsh[2][128 * 40];

  int lane = threadIdx.x & 31;
  int wv   = threadIdx.x >> 5;
  int wm   = wv & 1;
  int wn   = wv >> 1;
  long mblk = (long)blockIdx.x * 64;
  long nblk = (long)blockIdx.y * 128;

  int t    = threadIdx.x;
  int trow = t >> 2;          // 0..63
  int tseg = (t & 3) * 8;     // 0,8,16,24 (halfs within a 32-half row)

  const _Float16* agp  = A + (mblk + trow) * (long)K + tseg;
  const _Float16* wgp0 = W + (nblk + trow) * (long)K + tseg;
  const _Float16* wgp1 = W + (nblk + trow + 64) * (long)K + tseg;

  auto stage = [&](int buf, int k0) {
    async_cp16(&Ash[buf][trow * 40 + tseg],        agp  + k0);
    async_cp16(&Wsh[buf][trow * 40 + tseg],        wgp0 + k0);
    async_cp16(&Wsh[buf][(trow + 64) * 40 + tseg], wgp1 + k0);
  };

  v8f c[2][2] = {};
  int nIter = K / 32;
  stage(0, 0);
  for (int it = 0; it < nIter; ++it) {
    async_wait0();
    __syncthreads();
    if (it + 1 < nIter) stage((it + 1) & 1, (it + 1) * 32);

    const _Float16* ab = Ash[it & 1];
    const _Float16* wb = Wsh[it & 1];
    v16h a0 = load_afrag(ab, wm * 32,      0, 40, lane);
    v16h a1 = load_afrag(ab, wm * 32 + 16, 0, 40, lane);
    v16h b0 = load_bfrag(wb, wn * 32,      0, 40, lane);
    v16h b1 = load_bfrag(wb, wn * 32 + 16, 0, 40, lane);
    c[0][0] = wmma16(a0, b0, c[0][0]);
    c[0][1] = wmma16(a0, b1, c[0][1]);
    c[1][0] = wmma16(a1, b0, c[1][0]);
    c[1][1] = wmma16(a1, b1, c[1][1]);
  }

  // C/D layout: (vgpr r, lane): row = 16i + 8*(lane>>4) + r, col = 16j + (lane&15)
  long m0 = mblk + wm * 32;
  long n0 = nblk + wn * 32;
  int hi = lane >> 4, nn = lane & 15;
  for (int i = 0; i < 2; ++i)
    for (int j = 0; j < 2; ++j)
      for (int r = 0; r < 8; ++r) {
        long row = m0 + 16 * i + 8 * hi + r;
        long col = n0 + 16 * j + nn;
        float val = c[i][j][r] * outScale;
        if (HALF_OUT) Ch[row * (long)N + col] = (_Float16)val;
        else          Cf[row * (long)N + col] = val;
      }
}

// ---------------------------------------------------------------------------
// Flash attention: one block (4 waves, 128 thr) per (b, h, 64-row q block).
// qh already carries the 1/sqrt(D) scale. kh/vh/ah are [B*S, H*D] f16.
// K chunk staged via async global->LDS; V chunk transposed via VALU+ds.
// ---------------------------------------------------------------------------
__global__ __launch_bounds__(128) void attn_wmma(const _Float16* __restrict__ qh,
                                                 const _Float16* __restrict__ kh,
                                                 const _Float16* __restrict__ vh,
                                                 const unsigned char* __restrict__ mask,
                                                 _Float16* __restrict__ ah) {
  const int Nq = 1024, Nkv = 2048, HID = 1024;
  int lane = threadIdx.x & 31;
  int w    = threadIdx.x >> 5;           // wave 0..3, each owns 16 q rows
  int bx   = blockIdx.x;
  int qblk = bx & 15;                    // Nq/64 = 16 q blocks
  int h    = (bx >> 4) & 15;
  int b    = bx >> 8;
  long qrow0 = (long)b * Nq + qblk * 64 + w * 16;
  long krow0 = (long)b * Nkv;
  int hi = lane >> 4, nn = lane & 15;

  __shared__ alignas(16) _Float16 Ksh[64 * 72];     // K chunk [kv][d], ld 72
  __shared__ alignas(16) _Float16 Vt [64 * 72];     // V chunk transposed [d][kv]
  __shared__ alignas(16) _Float16 Psc[4][16 * 72];  // per-wave P scratch

  // Q tile 16x64 as two A-fragments, resident for the whole kv loop.
  v16h aq0 = load_afrag(qh, qrow0, h * 64,      HID, lane);
  v16h aq1 = load_afrag(qh, qrow0, h * 64 + 32, HID, lane);

  float Mrun[8], Lrun[8];
  v8f o[4] = {};
#pragma unroll
  for (int r = 0; r < 8; ++r) { Mrun[r] = -1e30f; Lrun[r] = 0.f; }

  for (int kv0 = 0; kv0 < Nkv; kv0 += 64) {
    __syncthreads();
    {   // stage K (async DMA) + V (transpose) chunk
      int t = threadIdx.x;
      int kvr = t >> 1, part = (t & 1) * 32;  // 64 rows x 2 half-rows
      long srow = (krow0 + kv0 + kvr) * (long)HID + h * 64 + part;
      const _Float16* ks = kh + srow;
      _Float16* kd = Ksh + kvr * 72 + part;
      async_cp16(kd + 0,  ks + 0);
      async_cp16(kd + 8,  ks + 8);
      async_cp16(kd + 16, ks + 16);
      async_cp16(kd + 24, ks + 24);

      const _Float16* vs = vh + srow;
      h4 vrow[8];
#pragma unroll
      for (int q8 = 0; q8 < 8; ++q8) vrow[q8] = *(const h4*)(vs + q8 * 4);
#pragma unroll
      for (int j = 0; j < 32; ++j)
        Vt[(part + j) * 72 + kvr] = vrow[j >> 2][j & 3];
    }
    async_wait0();
    __syncthreads();

    // scores S = Q * K^T : 4 tiles of 16x16, contraction K=64 in two WMMAs
    v8f s[4] = {};
#pragma unroll
    for (int t2 = 0; t2 < 4; ++t2) {
      v16h bk0 = load_bfrag(Ksh, t2 * 16, 0,  72, lane);
      s[t2] = wmma16(aq0, bk0, s[t2]);
      v16h bk1 = load_bfrag(Ksh, t2 * 16, 32, 72, lane);
      s[t2] = wmma16(aq1, bk1, s[t2]);
    }

    // mask + chunk row max
    float cmax[8];
#pragma unroll
    for (int r = 0; r < 8; ++r) cmax[r] = -1e30f;
#pragma unroll
    for (int t2 = 0; t2 < 4; ++t2) {
      bool mm = mask[(long)b * Nkv + kv0 + t2 * 16 + nn] != 0;
#pragma unroll
      for (int r = 0; r < 8; ++r) {
        float sv = mm ? -1e9f : (float)s[t2][r];
        s[t2][r] = sv;
        cmax[r] = fmaxf(cmax[r], sv);
      }
    }
#pragma unroll
    for (int off = 8; off >= 1; off >>= 1)
#pragma unroll
      for (int r = 0; r < 8; ++r)
        cmax[r] = fmaxf(cmax[r], __shfl_xor(cmax[r], off, 16));

    // online softmax update
    float alpha[8], rs[8];
#pragma unroll
    for (int r = 0; r < 8; ++r) {
      float nm = fmaxf(Mrun[r], cmax[r]);
      alpha[r] = __expf(Mrun[r] - nm);
      Mrun[r]  = nm;
      rs[r]    = 0.f;
    }
#pragma unroll
    for (int t2 = 0; t2 < 4; ++t2)
#pragma unroll
      for (int r = 0; r < 8; ++r) {
        float pv = __expf((float)s[t2][r] - Mrun[r]);
        rs[r] += pv;
        // C-layout -> A-layout via per-wave LDS scratch
        Psc[w][(8 * hi + r) * 72 + t2 * 16 + nn] = (_Float16)pv;
      }
#pragma unroll
    for (int off = 8; off >= 1; off >>= 1)
#pragma unroll
      for (int r = 0; r < 8; ++r)
        rs[r] += __shfl_xor(rs[r], off, 16);
#pragma unroll
    for (int r = 0; r < 8; ++r) Lrun[r] = Lrun[r] * alpha[r] + rs[r];
#pragma unroll
    for (int dt = 0; dt < 4; ++dt)
#pragma unroll
      for (int r = 0; r < 8; ++r)
        o[dt][r] = o[dt][r] * alpha[r];

    // O += P * V  (P from LDS as A-frags, V^T from LDS as B-frags)
#pragma unroll
    for (int kc = 0; kc < 64; kc += 32) {
      v16h ap = load_afrag(Psc[w], 0, kc, 72, lane);
#pragma unroll
      for (int dt = 0; dt < 4; ++dt) {
        v16h bv = load_bfrag(Vt, dt * 16, kc, 72, lane);
        o[dt] = wmma16(ap, bv, o[dt]);
      }
    }
  }

  // normalize and emit [B*Nq, H*D] f16 for the output projection GEMM
#pragma unroll
  for (int r = 0; r < 8; ++r) {
    float inv = 1.0f / Lrun[r];
    long row = (qrow0 + 8 * hi + r) * (long)HID + h * 64;
#pragma unroll
    for (int dt = 0; dt < 4; ++dt)
      ah[row + dt * 16 + nn] = (_Float16)(o[dt][r] * inv);
  }
}

// ---------------------------------------------------------------------------
// Launch
// ---------------------------------------------------------------------------
extern "C" void kernel_launch(void* const* d_in, const int* in_sizes, int n_in,
                              void* d_out, int out_size, void* d_ws, size_t ws_size,
                              hipStream_t stream) {
  (void)in_sizes; (void)n_in; (void)out_size; (void)ws_size;

  const float* vis = (const float*)d_in[0];
  const float* qry = (const float*)d_in[1];
  const unsigned char* mask = (const unsigned char*)d_in[2];
  const float* Wq = (const float*)d_in[3];
  const float* Wk = (const float*)d_in[4];
  const float* Wv = (const float*)d_in[5];
  const float* Wo = (const float*)d_in[6];

  const long B = 8, Nq = 1024, Nkv = 2048, QD = 1024, KD = 1152, HID = 1024;

  _Float16* p = (_Float16*)d_ws;
  _Float16* x16  = p; p += B * Nkv * KD;
  _Float16* q16  = p; p += B * Nq * QD;
  _Float16* wq16 = p; p += HID * QD;
  _Float16* wk16 = p; p += HID * KD;
  _Float16* wv16 = p; p += HID * KD;
  _Float16* wo16 = p; p += HID * HID;
  _Float16* qh   = p; p += B * Nq * HID;   // scaled Q projection
  _Float16* kh   = p; p += B * Nkv * HID;
  _Float16* vh   = p; p += B * Nkv * HID;
  _Float16* ah   = p; p += B * Nq * HID;

  cvt_f32_f16<<<2048, 256, 0, stream>>>(vis, x16, B * Nkv * KD);
  cvt_f32_f16<<<2048, 256, 0, stream>>>(qry, q16, B * Nq * QD);
  cvt_f32_f16<<<256, 256, 0, stream>>>(Wq, wq16, HID * QD);
  cvt_f32_f16<<<256, 256, 0, stream>>>(Wk, wk16, HID * KD);
  cvt_f32_f16<<<256, 256, 0, stream>>>(Wv, wv16, HID * KD);
  cvt_f32_f16<<<256, 256, 0, stream>>>(Wo, wo16, HID * HID);

  // Q projection with softmax scale folded in: 1/sqrt(64) = 0.125
  { dim3 g(8192 / 64, 1024 / 128);
    gemm_wmma<true><<<g, 256, 0, stream>>>(q16, wq16, nullptr, qh,
                                           8192, 1024, 1024, 0.125f); }
  { dim3 g(16384 / 64, 1024 / 128);
    gemm_wmma<true><<<g, 256, 0, stream>>>(x16, wk16, nullptr, kh,
                                           16384, 1024, 1152, 1.0f);
    gemm_wmma<true><<<g, 256, 0, stream>>>(x16, wv16, nullptr, vh,
                                           16384, 1024, 1152, 1.0f); }

  attn_wmma<<<8 * 16 * 16, 128, 0, stream>>>(qh, kh, vh, mask, ah);

  { dim3 g(8192 / 64, 1024 / 128);
    gemm_wmma<false><<<g, 256, 0, stream>>>(ah, wo16, (float*)d_out, nullptr,
                                            8192, 1024, 1024, 1.0f); }
}
